// AtomQuantizer_53661321396399
// MI455X (gfx1250) — matrix operand order repaired and checked by reading, not compiled, not run on hardware
//
#include <hip/hip_runtime.h>

typedef __attribute__((ext_vector_type(16))) _Float16 v16h;
typedef __attribute__((ext_vector_type(8)))  _Float16 v8h;
typedef __attribute__((ext_vector_type(4)))  _Float16 v4h;
typedef __attribute__((ext_vector_type(8)))  float    v8f;
typedef __attribute__((ext_vector_type(4)))  int      vi4;

#define NROWS 65536
#define DIM   512
#define KCB   4096

// slow path tiling
#define MT    128
#define JT    128
#define DT    128
#define LDH   136    // slow path LDS stride (128+8)
// fast path tiling
#define MT2   256
#define JT2   256
#define DT2   64
#define LDH2  72     // fast path LDS stride (64+8)
#define NCHUNK (DIM / DT2)   // 8

#define AS1 __attribute__((address_space(1)))
#define AS3 __attribute__((address_space(3)))

#if defined(__has_builtin)
#  if __has_builtin(__builtin_amdgcn_global_load_async_to_lds_b128)
#    define HAVE_ASYNC 1
#  endif
#  if __has_builtin(__builtin_amdgcn_s_wait_asynccnt)
#    define HAVE_WAIT_ASYNC 1
#  endif
#endif

__device__ __forceinline__ void wait_async_16() {
#if defined(HAVE_ASYNC)
#  if defined(HAVE_WAIT_ASYNC)
  __builtin_amdgcn_s_wait_asynccnt(16);
#  else
  asm volatile("s_wait_asynccnt 0x10" ::: "memory");
#  endif
#endif
}
__device__ __forceinline__ void wait_async_0() {
#if defined(HAVE_ASYNC)
#  if defined(HAVE_WAIT_ASYNC)
  __builtin_amdgcn_s_wait_asynccnt(0);
#  else
  asm volatile("s_wait_asynccnt 0x0" ::: "memory");
#  endif
#endif
}

__device__ __forceinline__ v16h concat8(v8h a, v8h b) {
  return __builtin_shufflevector(a, b, 0,1,2,3,4,5,6,7,8,9,10,11,12,13,14,15);
}

__device__ __forceinline__ void splitStore(float4 v, _Float16* hip_, _Float16* lop_) {
  v4h h, l;
  h.x = (_Float16)v.x; l.x = (_Float16)(v.x - (float)h.x);
  h.y = (_Float16)v.y; l.y = (_Float16)(v.y - (float)h.y);
  h.z = (_Float16)v.z; l.z = (_Float16)(v.z - (float)h.z);
  h.w = (_Float16)v.w; l.w = (_Float16)(v.w - (float)h.w);
  *(v4h*)hip_ = h; *(v4h*)lop_ = l;
}

// ---------------- kernel 0 (fast path): split f32 -> f16 hi/lo planes ----------------
__global__ __launch_bounds__(256)
void split_kernel(const float* __restrict__ src, _Float16* __restrict__ hi,
                  _Float16* __restrict__ lo) {
  size_t i = (size_t)blockIdx.x * blockDim.x + threadIdx.x;   // float4 index
  float4 v = *(const float4*)(src + i * 4);
  splitStore(v, hi + i * 4, lo + i * 4);
}

// ---------------- kernel 1: ||e_k||^2 ----------------
__global__ __launch_bounds__(256) void enorm_kernel(const float* __restrict__ emb,
                                                    float* __restrict__ enorm) {
  int lane = threadIdx.x & 31;
  int row  = (blockIdx.x * blockDim.x + threadIdx.x) >> 5;   // one wave per row
  const float4* p = (const float4*)(emb + (size_t)row * DIM);
  float s = 0.0f;
  #pragma unroll
  for (int i = 0; i < 4; ++i) {
    float4 v = p[lane + 32 * i];
    s += v.x*v.x + v.y*v.y + v.z*v.z + v.w*v.w;
  }
  #pragma unroll
  for (int off = 16; off >= 1; off >>= 1) s += __shfl_xor(s, off, 32);
  if (lane == 0) enorm[row] = s;
}

// ---------------- fast-path staging: 256x64 f16 plane chunk (async if available) ----------------
__device__ __forceinline__ void issue_plane(const _Float16* __restrict__ g, int gRowBase,
                                            int dBase, _Float16* s, int tid) {
  #pragma unroll
  for (int it = 0; it < 4; ++it) {
    int e8  = tid + it * 512;          // v8h index over 256x64 plane chunk (2048 total)
    int row = e8 >> 3;                 // 8 v8h per row
    int c8  = e8 & 7;
    const _Float16* src = g + (size_t)(gRowBase + row) * DIM + dBase + c8 * 8;
    _Float16* dst = s + row * LDH2 + c8 * 8;
#if defined(HAVE_ASYNC)
    __builtin_amdgcn_global_load_async_to_lds_b128(
        (AS1 vi4*)(uintptr_t)src,
        (AS3 vi4*)(unsigned)(uintptr_t)dst, 0, 0);
#else
    *(v8h*)dst = *(const v8h*)src;
#endif
  }
}

// ---------------- kernel 2 (fast): WMMA argmin, 256x256 tiles, double-buffered ----------------
__global__ __launch_bounds__(512)
void vq_argmin_fast(const _Float16* __restrict__ xh, const _Float16* __restrict__ xl,
                    const _Float16* __restrict__ eh, const _Float16* __restrict__ el,
                    const float* __restrict__ enorm, int* __restrict__ idxOut) {
  __shared__ __align__(16) _Float16 sAh[2][MT2 * LDH2];
  __shared__ __align__(16) _Float16 sAl[2][MT2 * LDH2];
  __shared__ __align__(16) _Float16 sBh[2][JT2 * LDH2];
  __shared__ __align__(16) _Float16 sBl[2][JT2 * LDH2];

  const int tid     = threadIdx.x;
  const int lane    = tid & 31;
  const int wave    = tid >> 5;            // 0..15
  const int l15     = lane & 15;
  const int halfSel = lane >> 4;
  const int rowBase = blockIdx.x * MT2;

  float mind[8];
  int   mini[8];
  #pragma unroll
  for (int r = 0; r < 8; ++r) { mind[r] = 3.4e38f; mini[r] = 0; }

  for (int jc = 0; jc < KCB / JT2; ++jc) {
    const int jBase = jc * JT2;
    v8f acc[16] = {};

    // prologue: stage chunk 0 into buffer 0
    issue_plane(xh, rowBase, 0, sAh[0], tid);
    issue_plane(xl, rowBase, 0, sAl[0], tid);
    issue_plane(eh, jBase,   0, sBh[0], tid);
    issue_plane(el, jBase,   0, sBl[0], tid);

    for (int dc = 0; dc < NCHUNK; ++dc) {
      const int cur = dc & 1;
      if (dc + 1 < NCHUNK) {
        const int nxt   = 1 - cur;
        const int dNext = (dc + 1) * DT2;
        issue_plane(xh, rowBase, dNext, sAh[nxt], tid);
        issue_plane(xl, rowBase, dNext, sAl[nxt], tid);
        issue_plane(eh, jBase,   dNext, sBh[nxt], tid);
        issue_plane(el, jBase,   dNext, sBl[nxt], tid);
        wait_async_16();          // in-order completion: current chunk's 16 loads done
      } else {
        wait_async_0();
      }
      __syncthreads();            // all waves' current chunk visible in LDS

      #pragma unroll
      for (int kk = 0; kk < DT2 / 32; ++kk) {
        const int k0 = kk * 32;
        const int aRow = wave * 16 + l15;
        const _Float16* pah = sAh[cur] + aRow * LDH2 + k0 + halfSel * 8;
        const _Float16* pal = sAl[cur] + aRow * LDH2 + k0 + halfSel * 8;
        v16h ah = concat8(*(const v8h*)pah, *(const v8h*)(pah + 16));
        v16h al = concat8(*(const v8h*)pal, *(const v8h*)(pal + 16));
        #pragma unroll
        for (int t = 0; t < 16; ++t) {
          const int bRow = t * 16 + l15;
          const _Float16* pbh = sBh[cur] + bRow * LDH2 + k0 + halfSel * 8;
          const _Float16* pbl = sBl[cur] + bRow * LDH2 + k0 + halfSel * 8;
          v16h bh = concat8(*(const v8h*)pbh, *(const v8h*)(pbh + 16));
          v16h bl = concat8(*(const v8h*)pbl, *(const v8h*)(pbl + 16));
          acc[t] = __builtin_amdgcn_wmma_f32_16x16x32_f16(false, ah, false, bh, (short)0, acc[t], false, false);
          acc[t] = __builtin_amdgcn_wmma_f32_16x16x32_f16(false, ah, false, bl, (short)0, acc[t], false, false);
          acc[t] = __builtin_amdgcn_wmma_f32_16x16x32_f16(false, al, false, bh, (short)0, acc[t], false, false);
        }
      }
      __syncthreads();            // all waves done reading before this buffer is re-staged
    }

    #pragma unroll
    for (int t = 0; t < 16; ++t) {
      const int n = jBase + t * 16 + l15;
      const float en = enorm[n];
      #pragma unroll
      for (int r = 0; r < 8; ++r) {
        float d = en - 2.0f * acc[t][r];
        if (d < mind[r]) { mind[r] = d; mini[r] = n; }
      }
    }
  }

  #pragma unroll
  for (int off = 8; off >= 1; off >>= 1) {
    #pragma unroll
    for (int r = 0; r < 8; ++r) {
      float od = __shfl_xor(mind[r], off, 32);
      int   oi = __shfl_xor(mini[r], off, 32);
      if (od < mind[r] || (od == mind[r] && oi < mini[r])) { mind[r] = od; mini[r] = oi; }
    }
  }
  if (l15 == 0) {
    #pragma unroll
    for (int r = 0; r < 8; ++r)
      idxOut[rowBase + wave * 16 + halfSel * 8 + r] = mini[r];
  }
}

// ---------------- kernel 2 (fallback): conversion-in-kernel variant ----------------
__global__ __launch_bounds__(256)
void vq_argmin_kernel(const float* __restrict__ x,
                      const float* __restrict__ emb,
                      const float* __restrict__ enorm,
                      int* __restrict__ idxOut) {
  __shared__ __align__(16) _Float16 sAh[MT * LDH];
  __shared__ __align__(16) _Float16 sAl[MT * LDH];
  __shared__ __align__(16) _Float16 sBh[JT * LDH];
  __shared__ __align__(16) _Float16 sBl[JT * LDH];

  const int tid     = threadIdx.x;
  const int lane    = tid & 31;
  const int wave    = tid >> 5;
  const int l15     = lane & 15;
  const int halfSel = lane >> 4;
  const int rowBase = blockIdx.x * MT;

  float mind[8];
  int   mini[8];
  #pragma unroll
  for (int r = 0; r < 8; ++r) { mind[r] = 3.4e38f; mini[r] = 0; }

  for (int jc = 0; jc < KCB / JT; ++jc) {
    const int jBase = jc * JT;
    v8f acc[8] = {};

    for (int dc = 0; dc < DIM / DT; ++dc) {
      const int dBase = dc * DT;
      __syncthreads();
      #pragma unroll
      for (int it = 0; it < 16; ++it) {
        int e4  = tid + it * 256;
        int row = e4 >> 5;
        int c4  = e4 & 31;
        float4 va = *(const float4*)(x   + (size_t)(rowBase + row) * DIM + dBase + c4 * 4);
        splitStore(va, sAh + row * LDH + c4 * 4, sAl + row * LDH + c4 * 4);
        float4 vb = *(const float4*)(emb + (size_t)(jBase   + row) * DIM + dBase + c4 * 4);
        splitStore(vb, sBh + row * LDH + c4 * 4, sBl + row * LDH + c4 * 4);
      }
      __syncthreads();

      #pragma unroll
      for (int kk = 0; kk < DT / 32; ++kk) {
        const int k0 = kk * 32;
        const int aRow = wave * 16 + l15;
        const _Float16* pah = sAh + aRow * LDH + k0 + halfSel * 8;
        const _Float16* pal = sAl + aRow * LDH + k0 + halfSel * 8;
        v16h ah = concat8(*(const v8h*)pah, *(const v8h*)(pah + 16));
        v16h al = concat8(*(const v8h*)pal, *(const v8h*)(pal + 16));
        #pragma unroll
        for (int t = 0; t < 8; ++t) {
          const int bRow = t * 16 + l15;
          const _Float16* pbh = sBh + bRow * LDH + k0 + halfSel * 8;
          const _Float16* pbl = sBl + bRow * LDH + k0 + halfSel * 8;
          v16h bh = concat8(*(const v8h*)pbh, *(const v8h*)(pbh + 16));
          v16h bl = concat8(*(const v8h*)pbl, *(const v8h*)(pbl + 16));
          acc[t] = __builtin_amdgcn_wmma_f32_16x16x32_f16(false, ah, false, bh, (short)0, acc[t], false, false);
          acc[t] = __builtin_amdgcn_wmma_f32_16x16x32_f16(false, ah, false, bl, (short)0, acc[t], false, false);
          acc[t] = __builtin_amdgcn_wmma_f32_16x16x32_f16(false, al, false, bh, (short)0, acc[t], false, false);
        }
      }
    }

    #pragma unroll
    for (int t = 0; t < 8; ++t) {
      const int n = jBase + t * 16 + l15;
      const float en = enorm[n];
      #pragma unroll
      for (int r = 0; r < 8; ++r) {
        float d = en - 2.0f * acc[t][r];
        if (d < mind[r]) { mind[r] = d; mini[r] = n; }
      }
    }
  }

  #pragma unroll
  for (int off = 8; off >= 1; off >>= 1) {
    #pragma unroll
    for (int r = 0; r < 8; ++r) {
      float od = __shfl_xor(mind[r], off, 32);
      int   oi = __shfl_xor(mini[r], off, 32);
      if (od < mind[r] || (od == mind[r] && oi < mini[r])) { mind[r] = od; mini[r] = oi; }
    }
  }
  if (l15 == 0) {
    #pragma unroll
    for (int r = 0; r < 8; ++r)
      idxOut[rowBase + wave * 16 + halfSel * 8 + r] = mini[r];
  }
}

// ---------------- kernel 3: gather + squared-error partials ----------------
__global__ __launch_bounds__(256)
void gather_loss_kernel(const float* __restrict__ x,
                        const float* __restrict__ emb,
                        const int* __restrict__ idx,
                        float* __restrict__ out,
                        float* __restrict__ partial) {
  __shared__ float ps[8];
  int lane = threadIdx.x & 31;
  int wave = threadIdx.x >> 5;
  int row  = blockIdx.x * 8 + wave;
  int k    = idx[row];
  const float4* q  = (const float4*)(emb + (size_t)k   * DIM);
  const float4* xr = (const float4*)(x   + (size_t)row * DIM);
  float4*       o  = (float4*)(out + (size_t)row * DIM);
  float s = 0.0f;
  #pragma unroll
  for (int i = 0; i < 4; ++i) {
    float4 qv = q[lane + 32 * i];
    float4 xv = xr[lane + 32 * i];
    float dx = qv.x - xv.x, dy = qv.y - xv.y, dz = qv.z - xv.z, dw = qv.w - xv.w;
    float4 ov;                                // straight-through forward: x + (q - x)
    ov.x = xv.x + dx; ov.y = xv.y + dy; ov.z = xv.z + dz; ov.w = xv.w + dw;
    o[lane + 32 * i] = ov;
    s += dx*dx + dy*dy + dz*dz + dw*dw;
  }
  #pragma unroll
  for (int off = 16; off >= 1; off >>= 1) s += __shfl_xor(s, off, 32);
  if (lane == 0) ps[wave] = s;
  __syncthreads();
  if (threadIdx.x == 0) {
    float t = 0.0f;
    #pragma unroll
    for (int i = 0; i < 8; ++i) t += ps[i];
    partial[blockIdx.x] = t;
  }
}

// ---------------- kernel 4: deterministic final reduction ----------------
__global__ __launch_bounds__(256)
void finalize_kernel(const float* __restrict__ partial, float* __restrict__ loss_out) {
  __shared__ float sm[256];
  float s = 0.0f;
  for (int i = threadIdx.x; i < NROWS / 8; i += 256) s += partial[i];
  sm[threadIdx.x] = s;
  __syncthreads();
  for (int off = 128; off >= 1; off >>= 1) {
    if (threadIdx.x < off) sm[threadIdx.x] += sm[threadIdx.x + off];
    __syncthreads();
  }
  if (threadIdx.x == 0)
    loss_out[0] = 1.25f * sm[0] / 33554432.0f;
}

extern "C" void kernel_launch(void* const* d_in, const int* in_sizes, int n_in,
                              void* d_out, int out_size, void* d_ws, size_t ws_size,
                              hipStream_t stream) {
  const float* x   = (const float*)d_in[0];
  const float* emb = (const float*)d_in[1];
  float* out = (float*)d_out;

  char* ws = (char*)d_ws;
  int*   idx   = (int*)ws;                                     // 256 KB
  float* enorm = (float*)(ws + (size_t)NROWS * 4);             // 16 KB
  float* part  = (float*)(ws + (size_t)NROWS * 4 + KCB * 4);   // 32 KB

  const size_t planesOff = 327680;                 // 320 KB, past small scratch
  const size_t xPlane = (size_t)NROWS * DIM * 2;   // 64 MB each
  const size_t ePlane = (size_t)KCB   * DIM * 2;   // 4 MB each
  const size_t need   = planesOff + 2 * xPlane + 2 * ePlane;

  enorm_kernel<<<KCB / 8, 256, 0, stream>>>(emb, enorm);

  if (ws_size >= need) {
    _Float16* xh = (_Float16*)(ws + planesOff);
    _Float16* xl = (_Float16*)(ws + planesOff + xPlane);
    _Float16* eh = (_Float16*)(ws + planesOff + 2 * xPlane);
    _Float16* el = (_Float16*)(ws + planesOff + 2 * xPlane + ePlane);
    split_kernel<<<(NROWS * (DIM / 4)) / 256, 256, 0, stream>>>(x, xh, xl);
    split_kernel<<<(KCB   * (DIM / 4)) / 256, 256, 0, stream>>>(emb, eh, el);
    vq_argmin_fast<<<NROWS / MT2, 512, 0, stream>>>(xh, xl, eh, el, enorm, idx);
  } else {
    vq_argmin_kernel<<<NROWS / MT, 256, 0, stream>>>(x, emb, enorm, idx);
  }

  gather_loss_kernel<<<NROWS / 8, 256, 0, stream>>>(x, emb, idx, out, part);
  finalize_kernel<<<1, 256, 0, stream>>>(part, out + (size_t)NROWS * DIM);
}
